// GCN2_523986010480
// MI455X (gfx1250) — matrix operand order repaired and checked by reading, not compiled, not run on hardware
//
#include <hip/hip_runtime.h>

// ---------------------------------------------------------------------------
// GCN x3 -> scalar mean for MI455X (gfx1250, wave32, WMMA).
// Stage 1: degree atomics (u32), Stage 2: edge scatter via native
// global_atomic_add_f32 (inline asm, guaranteed HW atomic into L2),
// Stage 3: bf16 WMMA projection with compile-time K=128 (4 unrolled
// v_wmma_f32_16x16x32_bf16), A-fragments register-resident across all
// N-tiles, fused bias+ReLU+global sum.
// ---------------------------------------------------------------------------

typedef __attribute__((ext_vector_type(16))) __bf16          v16bf;
typedef __attribute__((ext_vector_type(8)))  float           v8f;
typedef __attribute__((ext_vector_type(8)))  unsigned short  u16x8;

#define KSTEPS 4   // K = 128 = 4 x 32 (IN_F fixed by the reference)

__device__ __forceinline__ void gatomic_fadd_slow(float* p, float v) {
  __hip_atomic_fetch_add(p, v, __ATOMIC_RELAXED, __HIP_MEMORY_SCOPE_AGENT);
}

// Guaranteed native L2 float atomic (no CAS-loop fallback risk). Non-returning
// form; tracked by STOREcnt, drained by the implicit wait at s_endpgm.
__device__ __forceinline__ void gatomic_fadd(float* p, float v) {
  asm volatile("global_atomic_add_f32 %0, %1, off" :: "v"(p), "v"(v) : "memory");
}

__device__ __forceinline__ unsigned short f2bf_rne(float f) {
  unsigned u = __builtin_bit_cast(unsigned, f);
  u += 0x7FFFu + ((u >> 16) & 1u);   // round-to-nearest-even
  return (unsigned short)(u >> 16);
}

// -------------------------------- kernels ----------------------------------

__global__ void k_zero_u32(unsigned* __restrict__ p, long long n) {
  long long i = (long long)blockIdx.x * blockDim.x + threadIdx.x;
  long long s = (long long)gridDim.x * blockDim.x;
  for (; i < n; i += s) p[i] = 0u;
}

__global__ void k_degrees(const int* __restrict__ src, const int* __restrict__ dst,
                          unsigned* __restrict__ degout, unsigned* __restrict__ degin,
                          int E) {
  int i = blockIdx.x * blockDim.x + threadIdx.x;
  int s = gridDim.x * blockDim.x;
  for (; i < E; i += s) {
    atomicAdd(&degout[src[i]], 1u);
    atomicAdd(&degin[dst[i]], 1u);
  }
}

__global__ void k_invdeg(const unsigned* __restrict__ degout,
                         const unsigned* __restrict__ degin,
                         float* __restrict__ invout, float* __restrict__ invin, int N) {
  int i = blockIdx.x * blockDim.x + threadIdx.x;
  if (i < N) {
    invout[i] = rsqrtf(fmaxf((float)degout[i], 1.0f));
    invin[i]  = rsqrtf(fmaxf((float)degin[i],  1.0f));
  }
}

// One wave per edge; each lane moves 4 contiguous floats (coalesced float4
// gather, 4 native f32 atomic adds that resolve in L2).
__global__ void k_scatter(const float4* __restrict__ x4, const int* __restrict__ src,
                          const int* __restrict__ dst, const float* __restrict__ invout,
                          float* __restrict__ agg, int E, int INF) {
  int lane = threadIdx.x & 31;
  int e = blockIdx.x * (blockDim.x >> 5) + (threadIdx.x >> 5);
  if (e >= E) return;
  if (lane * 4 >= INF) return;
  int s = src[e], d = dst[e];
  float w = invout[s];
  float4 v = x4[(long long)s * (INF >> 2) + lane];
  float* ap = agg + (long long)d * INF + lane * 4;
  gatomic_fadd(ap + 0, v.x * w);
  gatomic_fadd(ap + 1, v.y * w);
  gatomic_fadd(ap + 2, v.z * w);
  gatomic_fadd(ap + 3, v.w * w);
}

// hn[i] = bf16(agg[i] * invin[row]);  INF is a power of two (128) -> shift.
__global__ void k_normcvt(const float* __restrict__ agg, const float* __restrict__ invin,
                          unsigned short* __restrict__ hn, long long total, int shift) {
  long long i = (long long)blockIdx.x * blockDim.x + threadIdx.x;
  long long s = (long long)gridDim.x * blockDim.x;
  for (; i < total; i += s)
    hn[i] = f2bf_rne(agg[i] * invin[i >> shift]);
}

// Pack W[K=INF][N=HID] (f32, row major) into per-wave B fragments (bf16):
//   Bp[ ((t*KSTEPS + s)*32 + lane)*16 + e ]  with  n = t*16 + (lane&15),
//   kb = lane<16 ? 0 : 16,  K = s*32 + kb + e      (contiguous k order, same as A)
__global__ void k_packB(const float* __restrict__ W, unsigned short* __restrict__ Bp,
                        int HID, long long total) {
  long long i = (long long)blockIdx.x * blockDim.x + threadIdx.x;
  if (i >= total) return;
  int e    = (int)(i & 15);
  int lane = (int)((i >> 4) & 31);
  long long rest = i >> 9;
  int s = (int)(rest % KSTEPS);
  int t = (int)(rest / KSTEPS);
  int n  = t * 16 + (lane & 15);
  int kb = (lane < 16) ? 0 : 16;
  int K  = s * 32 + kb + e;
  Bp[i] = f2bf_rne(W[(long long)K * HID + n]);
}

// One wave owns one M-tile: loads the 4 A-fragments (K=128) once into
// registers, then sweeps all N-tiles (B is 77 KB total -> L2/L0-hot).
// All 4 B fragments are loaded up front (one clause of 8 b128 loads) so the
// scheduler can overlap the next tile's loads with this tile's WMMAs.
__global__ void k_gemm_reduce(const unsigned short* __restrict__ hn,
                              const unsigned short* __restrict__ Bp,
                              const float* __restrict__ bias,
                              float* __restrict__ acc,
                              int Mtiles, int Ntiles, int INF) {
  int wave = threadIdx.x >> 5;
  int lane = threadIdx.x & 31;
  int mt = blockIdx.x * (blockDim.x >> 5) + wave;
  if (mt >= Mtiles) return;                 // wave-uniform -> EXEC stays all-ones

  int lm = lane & 15;
  // A fragment: lanes 0-15 row m, K base 0; lanes 16-31, K base 16 (contiguous).
  const unsigned short* arow = hn + ((long long)(mt * 16 + lm)) * INF;
  int kbA = (lane < 16) ? 0 : 16;

  union frag { u16x8 u[2]; v16bf v; };
  frag A[KSTEPS];
#pragma unroll
  for (int s = 0; s < KSTEPS; ++s) {
    A[s].u[0] = *(const u16x8*)(arow + s * 32 + kbA);       // K = s*32+kb .. +7
    A[s].u[1] = *(const u16x8*)(arow + s * 32 + 8 + kbA);   // K = s*32+8+kb .. +7
  }

  float sum = 0.f;
  for (int nt = 0; nt < Ntiles; ++nt) {
    const unsigned short* bp = Bp + ((long long)nt * KSTEPS * 32 + lane) * 16;
    frag B[KSTEPS];
#pragma unroll
    for (int s = 0; s < KSTEPS; ++s) {      // 8 x global_load_b128, clause-able
      B[s].u[0] = *(const u16x8*)(bp + s * 512);
      B[s].u[1] = *(const u16x8*)(bp + s * 512 + 8);
    }
    v8f c = {0.f, 0.f, 0.f, 0.f, 0.f, 0.f, 0.f, 0.f};
#pragma unroll
    for (int s = 0; s < KSTEPS; ++s) {
      c = __builtin_amdgcn_wmma_f32_16x16x32_bf16(
          /*neg_a=*/false, A[s].v, /*neg_b=*/false, B[s].v,
          /*c_mod=*/(short)0, c, /*reuse_a=*/false, /*reuse_b=*/false);
    }
    // C/D layout: VGPR r, lane l -> N = l&15 (bias is per-lane constant).
    float bn = bias[nt * 16 + lm];
#pragma unroll
    for (int r = 0; r < 8; ++r) {
      float v = c[r] + bn;
      sum += (v > 0.f) ? v : 0.f;
    }
  }

#pragma unroll
  for (int off = 16; off > 0; off >>= 1) sum += __shfl_xor(sum, off, 32);
  if (lane == 0) gatomic_fadd_slow(acc, sum);
}

__global__ void k_final(const float* __restrict__ acc, float* __restrict__ out,
                        float scale) {
  if (threadIdx.x == 0 && blockIdx.x == 0) out[0] = acc[0] * scale;
}

// ------------------------------- launcher ----------------------------------

static inline size_t alignup(size_t x) { return (x + 255) & ~(size_t)255; }

extern "C" void kernel_launch(void* const* d_in, const int* in_sizes, int n_in,
                              void* d_out, int out_size, void* d_ws, size_t ws_size,
                              hipStream_t stream) {
  const float* x[3]    = { (const float*)d_in[0], (const float*)d_in[1], (const float*)d_in[2] };
  const int*   srcs[3] = { (const int*)d_in[3], (const int*)d_in[5], (const int*)d_in[7] };
  const int*   dsts[3] = { (const int*)d_in[4], (const int*)d_in[6], (const int*)d_in[8] };
  const float* W    = (const float*)d_in[9];
  const float* bias = (const float*)d_in[10];

  const int HID = in_sizes[10];                 // 304
  const int INF = in_sizes[9] / HID;            // 128 (== KSTEPS*32)
  const int N   = in_sizes[0] / INF;            // 50000
  const int E   = in_sizes[3];                  // 800000
  const int Mtiles = N / 16;                    // 3125
  const int Ntiles = HID / 16;                  // 19
  int shift = 0; while ((1 << shift) < INF) ++shift;

  // scratch layout (reused serially across graphs): ~40 MB
  char* ws = (char*)d_ws;
  size_t off = 0;
  float*          acc    = (float*)(ws + off);          off = alignup(off + 256);
  unsigned short* Bp     = (unsigned short*)(ws + off); off = alignup(off + (size_t)Ntiles * KSTEPS * 32 * 16 * 2);
  unsigned*       degout = (unsigned*)(ws + off);       off = alignup(off + (size_t)N * 4);
  unsigned*       degin  = (unsigned*)(ws + off);       off = alignup(off + (size_t)N * 4);
  float*          invout = (float*)(ws + off);          off = alignup(off + (size_t)N * 4);
  float*          invin  = (float*)(ws + off);          off = alignup(off + (size_t)N * 4);
  float*          agg    = (float*)(ws + off);          off = alignup(off + (size_t)N * INF * 4);
  unsigned short* hn     = (unsigned short*)(ws + off); off = alignup(off + (size_t)N * INF * 2);
  (void)ws_size; (void)n_in; (void)out_size;

  const long long aggN   = (long long)N * INF;
  const long long bTotal = (long long)Ntiles * KSTEPS * 32 * 16;

  // accumulator + packed W (once)
  k_zero_u32<<<1, 64, 0, stream>>>((unsigned*)acc, 64);
  k_packB<<<(int)((bTotal + 255) / 256), 256, 0, stream>>>(W, Bp, HID, bTotal);

  for (int g = 0; g < 3; ++g) {
    // degout/degin are adjacent in ws -> one zero pass; agg separately.
    k_zero_u32<<<1024, 256, 0, stream>>>(degout, 2LL * N);
    k_zero_u32<<<2048, 256, 0, stream>>>((unsigned*)agg, aggN);

    k_degrees<<<2048, 256, 0, stream>>>(srcs[g], dsts[g], degout, degin, E);
    k_invdeg<<<(N + 255) / 256, 256, 0, stream>>>(degout, degin, invout, invin, N);

    // 8 edges per 256-thread block (one wave per edge)
    k_scatter<<<(E + 7) / 8, 256, 0, stream>>>((const float4*)x[g], srcs[g], dsts[g],
                                               invout, agg, E, INF);

    k_normcvt<<<2048, 256, 0, stream>>>(agg, invin, hn, aggN, shift);

    // one wave per M-tile, A-fragments register-resident across all N-tiles
    k_gemm_reduce<<<(Mtiles + 7) / 8, 256, 0, stream>>>(hn, Bp, bias, acc,
                                                        Mtiles, Ntiles, INF);
  }

  const float scale = 1.0f / (3.0f * (float)N * (float)HID);
  k_final<<<1, 32, 0, stream>>>(acc, (float*)d_out, scale);
}
